// AttentionDecoder_30013231464473
// MI455X (gfx1250) — matrix operand order repaired and checked by reading, not compiled, not run on hardware
//
#include <hip/hip_runtime.h>
#include <hip/hip_bf16.h>
#include <stdint.h>

// Problem constants (from reference setup_inputs)
#define B_   256
#define P_   256
#define ENC_ 256
#define DEC_ 512
#define E_   256
#define V_   5000
#define T_   96
#define XH_  1024      // E + ENC + DEC (fused LSTM input: [emb | gate*awe | h])
#define G4_  2048      // 4*DEC
#define NPAD_FC 5056   // V padded to multiple of 64 for the wave tile

typedef __attribute__((ext_vector_type(16))) __bf16        v16bf;
typedef __attribute__((ext_vector_type(8)))  float         v8f;
typedef __attribute__((ext_vector_type(8)))  unsigned int  v8u;

__device__ __forceinline__ float bf2f(unsigned short h) {
  return __uint_as_float(((unsigned int)h) << 16);
}
__device__ __forceinline__ unsigned short f2bf(float f) {
  unsigned int u = __float_as_uint(f);
  unsigned int r = ((u >> 16) & 1u) + 0x7fffu;   // round-to-nearest-even
  return (unsigned short)((u + r) >> 16);
}
__device__ __forceinline__ float sigf(float x) { return 1.f / (1.f + __expf(-x)); }

__device__ __forceinline__ v16bf load_frag(const unsigned short* p0,
                                           const unsigned short* p1) {
  uint4 a0 = *(const uint4*)p0;
  uint4 a1 = *(const uint4*)p1;
  v8u u; u[0]=a0.x; u[1]=a0.y; u[2]=a0.z; u[3]=a0.w;
         u[4]=a1.x; u[5]=a1.y; u[6]=a1.z; u[7]=a1.w;
  return __builtin_bit_cast(v16bf, u);
}

// ---------------------------------------------------------------------------
// Generic bf16 GEMM, C = A[M,K] * Bpacked[K,Npad] (+bias), f32 accumulate.
// A: row-major bf16. B: pre-swizzled into WMMA B-fragment order:
//   element idx = ((kc*NT + nt)*32 + lane)*16 + e  with
//   k = kc*32 + (lane>>4)*8 + (e&7) + (e>>3)*16,  n = nt*16 + (lane&15)
// Each wave computes a 32x64 tile: 2 A-fragments share 4 B-fragments per
// K=32 chunk -> 12 b128 loads per 8 WMMAs. Next-chunk B is prefetched at
// WGP scope so the L2->WGP$ fill happens a chunk ahead of use.
// mode 0: f32 out; mode 1: bf16 out; mode 2: f32 out with per-row length
//         mask and custom stride/offset (fc logits straight into d_out).
// ---------------------------------------------------------------------------
__global__ __launch_bounds__(256)
void gemm_bf16_wmma(const unsigned short* __restrict__ A,
                    const unsigned short* __restrict__ Bp,
                    const float* __restrict__ bias,
                    void* __restrict__ outp,
                    int M, int K, int Npad, int Ntrue,
                    long long ldo, long long out_off,
                    int mode, const int* __restrict__ lengths, int t)
{
  const int lane = threadIdx.x & 31;
  const int wave = threadIdx.x >> 5;
  const int tm = blockIdx.y * 64  + (wave & 1) * 32;   // M tile base (32 rows)
  const int tn = blockIdx.x * 256 + (wave >> 1) * 64;  // N tile base (64 cols)
  if (tm >= M || tn >= Npad) return;                   // no barriers below

  const int NT    = Npad >> 4;
  const int m0    = tm + (lane & 15);
  const int kbase = (lane >> 4) * 8;                   // A-fragment K origin
  const unsigned short* arow0 = A + (size_t)m0 * K + kbase;
  const unsigned short* arow1 = arow0 + (size_t)16 * K;
  const size_t bstep = (size_t)NT * 512;               // elements per K-chunk of Bp

  const v8f zero = {0.f,0.f,0.f,0.f,0.f,0.f,0.f,0.f};
  v8f acc[8];                                          // [0..3]: rows tm+0..15
#pragma unroll                                         // [4..7]: rows tm+16..31
  for (int j = 0; j < 8; ++j) acc[j] = zero;

  const int KC = K >> 5;
  for (int kc = 0; kc < KC; ++kc) {
    // Two A fragments (rows m0 and m0+16), ISA 16-bit A layout:
    // lane holds K = kbase+{0..7} and kbase+16+{0..7}
    v16bf af0 = load_frag(arow0 + kc * 32, arow0 + kc * 32 + 16);
    v16bf af1 = load_frag(arow1 + kc * 32, arow1 + kc * 32 + 16);
#pragma unroll
    for (int j = 0; j < 4; ++j) {
      const unsigned short* bptr =
          Bp + (((size_t)kc * NT + (tn >> 4) + j) * 32 + lane) * 16;
      // locality=3 -> WGP scope: pull into all cache levels (L2 -> WGP$)
      __builtin_prefetch((const void*)(bptr + bstep), 0, 3);
      v16bf bfv = load_frag(bptr, bptr + 8);
      acc[j] = __builtin_amdgcn_wmma_f32_16x16x32_bf16(
                   false, af0, false, bfv, (short)0, acc[j],     false, false);
      acc[4 + j] = __builtin_amdgcn_wmma_f32_16x16x32_bf16(
                   false, af1, false, bfv, (short)0, acc[4 + j], false, false);
    }
  }

  // Epilogue: lane L, vgpr r -> row base + (L>>4)*8 + r, col tn + (L&15)
  const int col0 = tn + (lane & 15);
#pragma unroll
  for (int g = 0; g < 2; ++g) {
    const int rbase = tm + g * 16 + (lane >> 4) * 8;
#pragma unroll
    for (int j = 0; j < 4; ++j) {
      int n = col0 + j * 16;
      if (n >= Ntrue) continue;
      float bv = bias ? bias[n] : 0.f;
#pragma unroll
      for (int r = 0; r < 8; ++r) {
        int mm = rbase + r;
        float v = acc[g * 4 + j][r] + bv;
        if (mode == 0) {
          ((float*)outp)[(size_t)mm * ldo + n] = v;
        } else if (mode == 1) {
          ((unsigned short*)outp)[(size_t)mm * ldo + n] = f2bf(v);
        } else {
          float act = (lengths[mm] > t) ? 1.f : 0.f;
          ((float*)outp)[(size_t)mm * ldo + out_off + n] = v * act;
        }
      }
    }
  }
}

// ---------------------------------------------------------------------------
// Pack an f32 weight matrix (optionally a concat of two) into the swizzled
// bf16 B-fragment layout. mode 0: single src. mode 1: N-concat at `split`.
// mode 2: K-concat at `split`. Pads n>=N with zeros.
// ---------------------------------------------------------------------------
__global__ void pack_b_kernel(const float* __restrict__ s0,
                              const float* __restrict__ s1,
                              unsigned short* __restrict__ dst,
                              int K, int N, int Npad, int mode, int split)
{
  long long idx = (long long)blockIdx.x * blockDim.x + threadIdx.x;
  long long total = (long long)K * Npad;
  if (idx >= total) return;
  int e    = (int)(idx & 15);
  int lane = (int)((idx >> 4) & 31);
  long long frag = idx >> 9;
  int NT = Npad >> 4;
  int nt = (int)(frag % NT);
  int kc = (int)(frag / NT);
  int k = kc * 32 + ((lane >> 4) << 3) + (e & 7) + ((e >> 3) << 4);
  int n = nt * 16 + (lane & 15);
  float v = 0.f;
  if (n < N) {
    if (mode == 0)      v = s0[(size_t)k * N + n];
    else if (mode == 1) v = (n < split) ? s0[(size_t)k * split + n]
                                        : s1[(size_t)k * (N - split) + (n - split)];
    else                v = (k < split) ? s0[(size_t)k * N + n]
                                        : s1[(size_t)(k - split) * N + n];
  }
  dst[idx] = f2bf(v);
}

// mode 0: dst = concat(s0[0:split], s1);  mode 1: dst = s0 + s1
__global__ void pack_bias_kernel(const float* __restrict__ s0,
                                 const float* __restrict__ s1,
                                 float* __restrict__ dst, int n, int split, int mode)
{
  int i = blockIdx.x * blockDim.x + threadIdx.x;
  if (i >= n) return;
  dst[i] = (mode == 0) ? ((i < split) ? s0[i] : s1[i - split]) : (s0[i] + s1[i]);
}

// Per-b mean over P (for h0/c0) fused with f32->bf16 cast of encoder_out.
__global__ __launch_bounds__(256)
void mean_and_cast_kernel(const float* __restrict__ enc,
                          unsigned short* __restrict__ enc_bf,
                          unsigned short* __restrict__ mean_bf)
{
  int b = blockIdx.x, e = threadIdx.x;
  float s = 0.f;
  for (int p = 0; p < P_; ++p) {
    float v = enc[((size_t)b * P_ + p) * ENC_ + e];
    enc_bf[((size_t)b * P_ + p) * ENC_ + e] = f2bf(v);
    s += v;
  }
  mean_bf[b * ENC_ + e] = f2bf(s * (1.0f / P_));
}

// hc0 [B,1024] -> h_bf16 [B,512], c [B,512]
__global__ void split_hc0_kernel(const float* __restrict__ hc0,
                                 unsigned short* __restrict__ h_bf,
                                 float* __restrict__ c)
{
  int idx = blockIdx.x * blockDim.x + threadIdx.x;   // B*DEC
  int b = idx >> 9, j = idx & (DEC_ - 1);
  h_bf[idx] = f2bf(hc0[(size_t)b * 1024 + j]);
  c[idx]    = hc0[(size_t)b * 1024 + DEC_ + j];
}

__global__ void write_lengths_kernel(const int* __restrict__ len,
                                     float* __restrict__ out)
{
  int i = threadIdx.x;
  out[i] = (float)len[i];
}

// Fused attention step (one block per batch row b):
//   e = relu(att1 + att2) . w_full   -> softmax over P -> alpha (written masked)
//   awe = sum_p alpha * enc;  gate = sigmoid(gate_pre);  build xh row in bf16.
__global__ __launch_bounds__(256)
void attn_step_kernel(const unsigned short* __restrict__ att1,   // [B*P, DEC] bf16
                      const float* __restrict__ att2g,           // [B, 768] = [att2 | gate_pre]
                      const unsigned short* __restrict__ enc_bf, // [B*P, ENC] bf16
                      const float* __restrict__ emb_w,           // [V, E]
                      const int*   __restrict__ captions,        // [B, T]
                      const float* __restrict__ faw,             // [DEC]
                      const float* __restrict__ fab,             // [1]
                      const unsigned short* __restrict__ h_bf,   // [B, DEC]
                      const int*   __restrict__ lengths,         // [B]
                      unsigned short* __restrict__ xh,           // [B, XH]
                      float* __restrict__ alphas_out,            // [B, T, P]
                      int t)
{
  __shared__ float att2s[DEC_];
  __shared__ float faws[DEC_];
  __shared__ float esc[P_];
  __shared__ float red[P_];
  __shared__ float als[P_];

  const int b   = blockIdx.x;
  const int tid = threadIdx.x;
  const int lane = tid & 31;
  const int wv   = tid >> 5;

  for (int d = tid; d < DEC_; d += 256) {
    att2s[d] = att2g[(size_t)b * 768 + d];
    faws[d]  = faw[d];
  }
  __syncthreads();

  const float fb = fab[0];
  // one wave per att1 row; coalesced b128 reads of bf16
  for (int p = wv; p < P_; p += 8) {
    const unsigned short* row = att1 + ((size_t)b * P_ + p) * DEC_;
    float s = 0.f;
#pragma unroll
    for (int c = 0; c < 2; ++c) {
      int d0 = c * 256 + lane * 8;
      uint4 x = *(const uint4*)(row + d0);
      unsigned int w[4] = {x.x, x.y, x.z, x.w};
#pragma unroll
      for (int q = 0; q < 4; ++q) {
        float lo = __uint_as_float(w[q] << 16);
        float hi = __uint_as_float(w[q] & 0xffff0000u);
        float r0 = fmaxf(lo + att2s[d0 + 2*q],     0.f);
        float r1 = fmaxf(hi + att2s[d0 + 2*q + 1], 0.f);
        s += r0 * faws[d0 + 2*q] + r1 * faws[d0 + 2*q + 1];
      }
    }
#pragma unroll
    for (int off = 16; off > 0; off >>= 1) s += __shfl_xor(s, off, 32);
    if (lane == 0) esc[p] = s + fb;
  }
  __syncthreads();

  // softmax over P=256
  red[tid] = esc[tid];
  __syncthreads();
  for (int s = 128; s > 0; s >>= 1) {
    if (tid < s) red[tid] = fmaxf(red[tid], red[tid + s]);
    __syncthreads();
  }
  float mx = red[0];
  __syncthreads();
  float ex = __expf(esc[tid] - mx);
  red[tid] = ex;
  __syncthreads();
  for (int s = 128; s > 0; s >>= 1) {
    if (tid < s) red[tid] += red[tid + s];
    __syncthreads();
  }
  float alpha = ex / red[0];
  als[tid] = alpha;

  float act = (lengths[b] > t) ? 1.f : 0.f;
  alphas_out[((size_t)b * T_ + t) * P_ + tid] = alpha * act;
  __syncthreads();

  // awe: thread tid owns channel e=tid; reads coalesced across threads
  float s = 0.f;
  const unsigned short* ecol = enc_bf + (size_t)b * P_ * ENC_ + tid;
  for (int p = 0; p < P_; ++p)
    s += als[p] * bf2f(ecol[(size_t)p * ENC_]);

  float gate = sigf(att2g[(size_t)b * 768 + DEC_ + tid]);

  unsigned short* xrow = xh + (size_t)b * XH_;
  int cap = captions[b * T_ + t];
  xrow[tid]       = f2bf(emb_w[(size_t)cap * E_ + tid]);   // embedding gather
  xrow[E_ + tid]  = f2bf(gate * s);                         // gated attention
  xrow[512 + tid] = h_bf[(size_t)b * DEC_ + tid];           // h (for W_hh part)
  xrow[768 + tid] = h_bf[(size_t)b * DEC_ + 256 + tid];
}

// LSTM cell pointwise: torch gate order i,f,g,o
__global__ void lstm_pointwise_kernel(const float* __restrict__ gates, // [B, 4D]
                                      float* __restrict__ c,           // [B, D]
                                      unsigned short* __restrict__ h_bf)
{
  int idx = blockIdx.x * blockDim.x + threadIdx.x;   // B*DEC
  int b = idx >> 9, j = idx & (DEC_ - 1);
  const float* g = gates + (size_t)b * G4_;
  float i_ = sigf(g[j]);
  float f_ = sigf(g[DEC_ + j]);
  float gg = tanhf(g[2 * DEC_ + j]);
  float o_ = sigf(g[3 * DEC_ + j]);
  float cn = f_ * c[idx] + i_ * gg;
  c[idx] = cn;
  h_bf[idx] = f2bf(o_ * tanhf(cn));
}

// ---------------------------------------------------------------------------
extern "C" void kernel_launch(void* const* d_in, const int* in_sizes, int n_in,
                              void* d_out, int out_size, void* d_ws, size_t ws_size,
                              hipStream_t stream)
{
  (void)in_sizes; (void)n_in; (void)out_size; (void)ws_size;
  const float* encoder_out = (const float*)d_in[0];
  const int*   captions    = (const int*)d_in[1];
  const int*   lengths     = (const int*)d_in[2];
  const float* embedding_w = (const float*)d_in[3];
  const float* enc_att_w   = (const float*)d_in[4];
  const float* enc_att_b   = (const float*)d_in[5];
  const float* dec_att_w   = (const float*)d_in[6];
  const float* dec_att_b   = (const float*)d_in[7];
  const float* full_att_w  = (const float*)d_in[8];
  const float* full_att_b  = (const float*)d_in[9];
  const float* lstm_wih    = (const float*)d_in[10];
  const float* lstm_whh    = (const float*)d_in[11];
  const float* lstm_bih    = (const float*)d_in[12];
  const float* lstm_bhh    = (const float*)d_in[13];
  const float* init_h_w    = (const float*)d_in[14];
  const float* init_h_b    = (const float*)d_in[15];
  const float* init_c_w    = (const float*)d_in[16];
  const float* init_c_b    = (const float*)d_in[17];
  const float* f_beta_w    = (const float*)d_in[18];
  const float* f_beta_b    = (const float*)d_in[19];
  const float* fc_w        = (const float*)d_in[20];
  const float* fc_b        = (const float*)d_in[21];

  float* out = (float*)d_out;

  // -------- carve workspace (256B aligned regions) --------
  char* p = (char*)d_ws;
  auto carve = [&](size_t bytes) -> void* {
    char* r = p; p += (bytes + 255) & ~(size_t)255; return (void*)r;
  };
  unsigned short* enc_bf   = (unsigned short*)carve((size_t)B_*P_*ENC_*2);   // 33.5 MB
  unsigned short* att1     = (unsigned short*)carve((size_t)B_*P_*DEC_*2);   // 67 MB
  unsigned short* wcomb_p  = (unsigned short*)carve((size_t)DEC_*768*2);
  unsigned short* wlstm_p  = (unsigned short*)carve((size_t)XH_*G4_*2);
  unsigned short* fc_p     = (unsigned short*)carve((size_t)DEC_*NPAD_FC*2);
  unsigned short* init_p   = (unsigned short*)carve((size_t)ENC_*1024*2);
  unsigned short* encatt_p = (unsigned short*)carve((size_t)ENC_*DEC_*2);
  float* bias1   = (float*)carve(768*4);
  float* biasl   = (float*)carve((size_t)G4_*4);
  float* biasi   = (float*)carve(1024*4);
  unsigned short* mean_bf = (unsigned short*)carve((size_t)B_*ENC_*2);
  float* hc0     = (float*)carve((size_t)B_*1024*4);
  unsigned short* h_bf = (unsigned short*)carve((size_t)B_*DEC_*2);
  float* cst     = (float*)carve((size_t)B_*DEC_*4);
  float* att2g   = (float*)carve((size_t)B_*768*4);
  unsigned short* xh = (unsigned short*)carve((size_t)B_*XH_*2);
  float* gates   = (float*)carve((size_t)B_*G4_*4);

  const size_t pred_sz = (size_t)B_ * T_ * V_;
  float* len_out   = out + pred_sz;
  float* alpha_out = len_out + B_;

  auto packGrid = [](long long total) {
    return dim3((unsigned)((total + 255) / 256));
  };

  // -------- one-time: pack weights/biases to bf16 fragment layout --------
  pack_b_kernel<<<packGrid((long long)DEC_*768),     256, 0, stream>>>(dec_att_w, f_beta_w, wcomb_p, DEC_, 768, 768, 1, DEC_);
  pack_b_kernel<<<packGrid((long long)XH_*G4_),      256, 0, stream>>>(lstm_wih, lstm_whh, wlstm_p, XH_, G4_, G4_, 2, 512);
  pack_b_kernel<<<packGrid((long long)DEC_*NPAD_FC), 256, 0, stream>>>(fc_w, nullptr, fc_p, DEC_, V_, NPAD_FC, 0, 0);
  pack_b_kernel<<<packGrid((long long)ENC_*1024),    256, 0, stream>>>(init_h_w, init_c_w, init_p, ENC_, 1024, 1024, 1, DEC_);
  pack_b_kernel<<<packGrid((long long)ENC_*DEC_),    256, 0, stream>>>(enc_att_w, nullptr, encatt_p, ENC_, DEC_, DEC_, 0, 0);
  pack_bias_kernel<<<3, 256, 0, stream>>>(dec_att_b, f_beta_b, bias1, 768, DEC_, 0);
  pack_bias_kernel<<<8, 256, 0, stream>>>(lstm_bih, lstm_bhh, biasl, G4_, 0, 1);
  pack_bias_kernel<<<4, 256, 0, stream>>>(init_h_b, init_c_b, biasi, 1024, DEC_, 0);
  write_lengths_kernel<<<1, 256, 0, stream>>>(lengths, len_out);
  mean_and_cast_kernel<<<B_, 256, 0, stream>>>(encoder_out, enc_bf, mean_bf);

  // att1 = enc @ enc_att_w + b   (stored bf16, reread every step)
  gemm_bf16_wmma<<<dim3(DEC_/256, (B_*P_)/64), 256, 0, stream>>>(
      enc_bf, encatt_p, enc_att_b, att1, B_*P_, ENC_, DEC_, DEC_,
      (long long)DEC_, 0, 1, nullptr, 0);
  // [h0 | c0] = mean_enc @ [init_h_w | init_c_w] + bias
  gemm_bf16_wmma<<<dim3(1024/256, B_/64), 256, 0, stream>>>(
      mean_bf, init_p, biasi, hc0, B_, ENC_, 1024, 1024,
      (long long)1024, 0, 0, nullptr, 0);
  split_hc0_kernel<<<(B_*DEC_)/256, 256, 0, stream>>>(hc0, h_bf, cst);

  // -------- sequential decode: 96 steps --------
  for (int t = 0; t < T_; ++t) {
    // [att2 | gate_pre] = h @ [dec_att_w | f_beta_w] + [dec_att_b | f_beta_b]
    gemm_bf16_wmma<<<dim3(768/256, B_/64), 256, 0, stream>>>(
        h_bf, wcomb_p, bias1, att2g, B_, DEC_, 768, 768,
        (long long)768, 0, 0, nullptr, 0);
    attn_step_kernel<<<B_, 256, 0, stream>>>(
        att1, att2g, enc_bf, embedding_w, captions, full_att_w, full_att_b,
        h_bf, lengths, xh, alpha_out, t);
    // gates = [emb | gate*awe | h] @ [W_ih ; W_hh] + (b_ih + b_hh)
    gemm_bf16_wmma<<<dim3(G4_/256, B_/64), 256, 0, stream>>>(
        xh, wlstm_p, biasl, gates, B_, XH_, G4_, G4_,
        (long long)G4_, 0, 0, nullptr, 0);
    lstm_pointwise_kernel<<<(B_*DEC_)/256, 256, 0, stream>>>(gates, cst, h_bf);
    // preds[:, t, :] = mask * (h_new @ fc_w + fc_b), written into d_out
    gemm_bf16_wmma<<<dim3((NPAD_FC+255)/256, B_/64), 256, 0, stream>>>(
        h_bf, fc_p, fc_b, out, B_, DEC_, NPAD_FC, V_,
        (long long)T_*V_, (long long)t*V_, 2, lengths, t);
  }
}
